// MultiAttentionHead_90151363543562
// MI455X (gfx1250) — matrix operand order repaired and checked by reading, not compiled, not run on hardware
//
#include <hip/hip_runtime.h>

// ---------------------------------------------------------------------------
// Problem constants (match the reference)
// ---------------------------------------------------------------------------
constexpr int Bc  = 4;
constexpr int Sc  = 1024;
constexpr int SKc = 1024;
constexpr int Gc  = 16;
constexpr int DQc = 64;
constexpr int DVc = 64;
constexpr int NFc = 1024;          // G*DQ == G*DV == Q_IN == OUT
constexpr float EPSc = 1e-5f;

// ---------------------------------------------------------------------------
// Types for WMMA fragments (gfx1250 wave32)
// ---------------------------------------------------------------------------
typedef __bf16 bf16x16 __attribute__((ext_vector_type(16)));
typedef float  f32x8   __attribute__((ext_vector_type(8)));

struct alignas(16) u32x4 { unsigned int x, y, z, w; };
struct alignas(16) f32x4 { float x, y, z, w; };

__device__ __forceinline__ unsigned short f32_to_bf16(float f) {
    unsigned int u = __float_as_uint(f);
    u = u + 0x7FFFu + ((u >> 16) & 1u);   // round-to-nearest-even
    return (unsigned short)(u >> 16);
}

// A-matrix 16x32 bf16 fragment.
// Lanes 0-15: row = lane, K = {k0..k0+7} and {k0+16..k0+23}
// Lanes 16-31: row = lane-16, K = {k0+8..k0+15} and {k0+24..k0+31}
__device__ __forceinline__ bf16x16
load_frag_a_bf16(const unsigned short* __restrict__ rowbase, int k0, int lane) {
    const int aksel = (lane >> 4) * 8;
    union { u32x4 q[2]; bf16x16 v; } f;
    f.q[0] = *reinterpret_cast<const u32x4*>(rowbase + k0 + aksel);
    f.q[1] = *reinterpret_cast<const u32x4*>(rowbase + k0 + 16 + aksel);
    return f.v;
}

// Same fragment but source data is f32 (softmax probs) -> convert in-register.
__device__ __forceinline__ bf16x16
load_frag_a_f32(const float* __restrict__ rowbase, int k0, int lane) {
    const int aksel = (lane >> 4) * 8;
    f32x4 c0 = *reinterpret_cast<const f32x4*>(rowbase + k0 + aksel);
    f32x4 c1 = *reinterpret_cast<const f32x4*>(rowbase + k0 + aksel + 4);
    f32x4 c2 = *reinterpret_cast<const f32x4*>(rowbase + k0 + 16 + aksel);
    f32x4 c3 = *reinterpret_cast<const f32x4*>(rowbase + k0 + 16 + aksel + 4);
    union { unsigned short s[16]; bf16x16 v; } f;
    f.s[0]  = f32_to_bf16(c0.x); f.s[1]  = f32_to_bf16(c0.y);
    f.s[2]  = f32_to_bf16(c0.z); f.s[3]  = f32_to_bf16(c0.w);
    f.s[4]  = f32_to_bf16(c1.x); f.s[5]  = f32_to_bf16(c1.y);
    f.s[6]  = f32_to_bf16(c1.z); f.s[7]  = f32_to_bf16(c1.w);
    f.s[8]  = f32_to_bf16(c2.x); f.s[9]  = f32_to_bf16(c2.y);
    f.s[10] = f32_to_bf16(c2.z); f.s[11] = f32_to_bf16(c2.w);
    f.s[12] = f32_to_bf16(c3.x); f.s[13] = f32_to_bf16(c3.y);
    f.s[14] = f32_to_bf16(c3.z); f.s[15] = f32_to_bf16(c3.w);
    return f.v;
}

// B-matrix 32x16 bf16 fragment read from a [N,K] row-major "weight" layout:
// lane's column = N index, lanes 0-15 hold K {k0..k0+7} and {k0+16..k0+23},
// lanes 16-31 hold K {k0+8..k0+15} and {k0+24..k0+31}.
__device__ __forceinline__ bf16x16
load_frag_b_bf16(const unsigned short* __restrict__ colbase, int k0, int lane) {
    const int bksel = (lane >> 4) * 8;
    union { u32x4 q[2]; bf16x16 v; } f;
    f.q[0] = *reinterpret_cast<const u32x4*>(colbase + k0 + bksel);
    f.q[1] = *reinterpret_cast<const u32x4*>(colbase + k0 + 16 + bksel);
    return f.v;
}

__device__ __forceinline__ f32x8 wmma_bf16(bf16x16 a, bf16x16 b, f32x8 c) {
    return __builtin_amdgcn_wmma_f32_16x16x32_bf16(
        /*neg_a=*/false, a, /*neg_b=*/false, b,
        /*c_mod=*/(short)0, c, /*reuse_a=*/false, /*reuse_b=*/false);
}

// A fragment set for a 32(M) x 64(N) wave tile at one k-step.
struct FragSet { bf16x16 a0, a1, b0, b1, b2, b3; };

__device__ __forceinline__ void
load_set_bf16(FragSet& s,
              const unsigned short* __restrict__ arow0,
              const unsigned short* __restrict__ arow1,
              const unsigned short* __restrict__ w0,
              const unsigned short* __restrict__ w1,
              const unsigned short* __restrict__ w2,
              const unsigned short* __restrict__ w3,
              int k0, int lane) {
    s.a0 = load_frag_a_bf16(arow0, k0, lane);
    s.a1 = load_frag_a_bf16(arow1, k0, lane);
    s.b0 = load_frag_b_bf16(w0, k0, lane);
    s.b1 = load_frag_b_bf16(w1, k0, lane);
    s.b2 = load_frag_b_bf16(w2, k0, lane);
    s.b3 = load_frag_b_bf16(w3, k0, lane);
}

__device__ __forceinline__ void
load_set_f32a(FragSet& s,
              const float* __restrict__ arow0,
              const float* __restrict__ arow1,
              const unsigned short* __restrict__ w0,
              const unsigned short* __restrict__ w1,
              const unsigned short* __restrict__ w2,
              const unsigned short* __restrict__ w3,
              int k0, int lane) {
    s.a0 = load_frag_a_f32(arow0, k0, lane);
    s.a1 = load_frag_a_f32(arow1, k0, lane);
    s.b0 = load_frag_b_bf16(w0, k0, lane);
    s.b1 = load_frag_b_bf16(w1, k0, lane);
    s.b2 = load_frag_b_bf16(w2, k0, lane);
    s.b3 = load_frag_b_bf16(w3, k0, lane);
}

__device__ __forceinline__ void
compute_set(const FragSet& s, f32x8* acc0, f32x8* acc1) {
    acc0[0] = wmma_bf16(s.a0, s.b0, acc0[0]);
    acc1[0] = wmma_bf16(s.a1, s.b0, acc1[0]);
    acc0[1] = wmma_bf16(s.a0, s.b1, acc0[1]);
    acc1[1] = wmma_bf16(s.a1, s.b1, acc1[1]);
    acc0[2] = wmma_bf16(s.a0, s.b2, acc0[2]);
    acc1[2] = wmma_bf16(s.a1, s.b2, acc1[2]);
    acc0[3] = wmma_bf16(s.a0, s.b3, acc0[3]);
    acc1[3] = wmma_bf16(s.a1, s.b3, acc1[3]);
}

// ---------------------------------------------------------------------------
// fp32 -> bf16 conversion
// ---------------------------------------------------------------------------
__global__ void cvt_f32_bf16(const float* __restrict__ src,
                             unsigned short* __restrict__ dst, int n) {
    int i = blockIdx.x * 256 + threadIdx.x;
    if (i < n) dst[i] = f32_to_bf16(src[i]);
}

// ---------------------------------------------------------------------------
// Generic GEMM: C[M,N] = A[M,K] * W[N,K]^T + bias (+ residual)
// block = 256 threads (8 waves); block tile = 256(M) x 64(N);
// wave tile = 32(M) x 64(N) -> 8 accumulators, 8 WMMA per 12 b128 loads.
// Ping-pong double buffer over k (no rotation copies); K % 64 == 0.
// launch_bounds(256,2): >=2 waves/SIMD => VGPR cap ~512, avoids spills.
// ---------------------------------------------------------------------------
__global__ void __launch_bounds__(256, 2)
gemm_wmma(const unsigned short* __restrict__ A, int lda,
          const unsigned short* __restrict__ W, int ldw,
          const float* __restrict__ bias,
          const float* __restrict__ residual,
          unsigned short* __restrict__ outB,
          float* __restrict__ outF,
          int N, int K) {
    const int lane  = threadIdx.x & 31;
    const int wave  = threadIdx.x >> 5;
    const int tileM = blockIdx.y * 256 + wave * 32;
    const int tileN = blockIdx.x * 64;
    const int lm    = lane & 15;

    const unsigned short* arow0 = A + (size_t)(tileM + lm) * lda;
    const unsigned short* arow1 = A + (size_t)(tileM + 16 + lm) * lda;
    const unsigned short* wcol0 = W + (size_t)(tileN + 0 * 16 + lm) * ldw;
    const unsigned short* wcol1 = W + (size_t)(tileN + 1 * 16 + lm) * ldw;
    const unsigned short* wcol2 = W + (size_t)(tileN + 2 * 16 + lm) * ldw;
    const unsigned short* wcol3 = W + (size_t)(tileN + 3 * 16 + lm) * ldw;

    f32x8 acc0[4] = {f32x8{}, f32x8{}, f32x8{}, f32x8{}};
    f32x8 acc1[4] = {f32x8{}, f32x8{}, f32x8{}, f32x8{}};

    FragSet s0, s1;
    load_set_bf16(s0, arow0, arow1, wcol0, wcol1, wcol2, wcol3, 0, lane);

    int k0 = 0;
    for (; k0 + 64 < K; k0 += 64) {
        __builtin_prefetch(arow0 + k0 + 96, 0, 3);
        load_set_bf16(s1, arow0, arow1, wcol0, wcol1, wcol2, wcol3, k0 + 32, lane);
        compute_set(s0, acc0, acc1);
        load_set_bf16(s0, arow0, arow1, wcol0, wcol1, wcol2, wcol3, k0 + 64, lane);
        compute_set(s1, acc0, acc1);
    }
    // tail: two k-steps remain (k0 and k0+32)
    load_set_bf16(s1, arow0, arow1, wcol0, wcol1, wcol2, wcol3, k0 + 32, lane);
    compute_set(s0, acc0, acc1);
    compute_set(s1, acc0, acc1);

    const int rb0 = tileM + ((lane >> 4) << 3);
#pragma unroll
    for (int n = 0; n < 4; ++n) {
        const int c = tileN + n * 16 + lm;
        const float bv = bias ? bias[c] : 0.0f;
#pragma unroll
        for (int j = 0; j < 8; ++j) {
            const float v0 = acc0[n][j] + bv;
            const float v1 = acc1[n][j] + bv;
            const size_t i0 = (size_t)(rb0 + j) * N + c;
            const size_t i1 = (size_t)(rb0 + 16 + j) * N + c;
            if (outF) {
                outF[i0] = residual ? v0 + residual[i0] : v0;
                outF[i1] = residual ? v1 + residual[i1] : v1;
            } else {
                outB[i0] = f32_to_bf16(v0);
                outB[i1] = f32_to_bf16(v1);
            }
        }
    }
}

// ---------------------------------------------------------------------------
// v transpose: vb[b, t, d*G + g] (bf16) -> vT[(b*G+g)*DV + d, t] (bf16)
// ---------------------------------------------------------------------------
__global__ void transpose_v(const unsigned short* __restrict__ vb,
                            unsigned short* __restrict__ vT) {
    const size_t i = (size_t)blockIdx.x * 256 + threadIdx.x;   // over B*G*DV*SK
    const int t = (int)(i & (SKc - 1));
    const size_t rest = i >> 10;
    const int d  = (int)(rest & (DVc - 1));
    const size_t bg = rest >> 6;
    const int b = (int)(bg >> 4);
    const int g = (int)(bg & (Gc - 1));
    vT[i] = vb[((size_t)(b * SKc + t)) * NFc + d * Gc + g];
}

// ---------------------------------------------------------------------------
// logits[bg, s, t] = scale * sum_d q[b,s,g*64+d] * k[b,t,g*64+d]
// grid = (SK/64, S/128, B*G); K = 64 (two k-steps, both load groups hoisted)
// ---------------------------------------------------------------------------
__global__ void __launch_bounds__(256, 2)
attn_logits(const unsigned short* __restrict__ qb,
            const unsigned short* __restrict__ kb,
            float* __restrict__ scores, float scale) {
    const int lane  = threadIdx.x & 31;
    const int wave  = threadIdx.x >> 5;
    const int lm    = lane & 15;
    const int bg    = blockIdx.z;
    const int b     = bg >> 4;
    const int g     = bg & (Gc - 1);
    const int tileS = blockIdx.y * 128 + wave * 16;
    const int tileT = blockIdx.x * 64;

    const unsigned short* qrow =
        qb + ((size_t)(b * Sc + tileS + lm)) * NFc + g * DQc;
    const unsigned short* kr0 =
        kb + ((size_t)(b * SKc + tileT + 0 * 16 + lm)) * NFc + g * DQc;
    const unsigned short* kr1 =
        kb + ((size_t)(b * SKc + tileT + 1 * 16 + lm)) * NFc + g * DQc;
    const unsigned short* kr2 =
        kb + ((size_t)(b * SKc + tileT + 2 * 16 + lm)) * NFc + g * DQc;
    const unsigned short* kr3 =
        kb + ((size_t)(b * SKc + tileT + 3 * 16 + lm)) * NFc + g * DQc;

    // hoist both k-steps' fragments so the first WMMA round overlaps the
    // second round's loads
    bf16x16 aA = load_frag_a_bf16(qrow, 0, lane);
    bf16x16 aB = load_frag_a_bf16(qrow, 32, lane);
    bf16x16 b0A = load_frag_b_bf16(kr0, 0, lane);
    bf16x16 b1A = load_frag_b_bf16(kr1, 0, lane);
    bf16x16 b2A = load_frag_b_bf16(kr2, 0, lane);
    bf16x16 b3A = load_frag_b_bf16(kr3, 0, lane);
    bf16x16 b0B = load_frag_b_bf16(kr0, 32, lane);
    bf16x16 b1B = load_frag_b_bf16(kr1, 32, lane);
    bf16x16 b2B = load_frag_b_bf16(kr2, 32, lane);
    bf16x16 b3B = load_frag_b_bf16(kr3, 32, lane);

    f32x8 acc[4] = {f32x8{}, f32x8{}, f32x8{}, f32x8{}};
    acc[0] = wmma_bf16(aA, b0A, acc[0]);
    acc[1] = wmma_bf16(aA, b1A, acc[1]);
    acc[2] = wmma_bf16(aA, b2A, acc[2]);
    acc[3] = wmma_bf16(aA, b3A, acc[3]);
    acc[0] = wmma_bf16(aB, b0B, acc[0]);
    acc[1] = wmma_bf16(aB, b1B, acc[1]);
    acc[2] = wmma_bf16(aB, b2B, acc[2]);
    acc[3] = wmma_bf16(aB, b3B, acc[3]);

    const int rbase = tileS + ((lane >> 4) << 3);
#pragma unroll
    for (int n = 0; n < 4; ++n) {
        const int c = tileT + n * 16 + lm;
#pragma unroll
        for (int j = 0; j < 8; ++j) {
            const int r = rbase + j;
            scores[((size_t)bg * Sc + r) * SKc + c] = acc[n][j] * scale;
        }
    }
}

// ---------------------------------------------------------------------------
// In-place softmax over rows of length 1024; one block (256 threads) per row.
// ---------------------------------------------------------------------------
__global__ void softmax_rows(float* __restrict__ scores) {
    __shared__ float red[256];
    const size_t row = blockIdx.x;
    float* p = scores + row * (size_t)SKc;
    const int t = threadIdx.x;

    float x[4];
    float m = -3.402823466e+38f;
#pragma unroll
    for (int i = 0; i < 4; ++i) {
        x[i] = p[t + i * 256];
        m = fmaxf(m, x[i]);
    }
    red[t] = m;
    __syncthreads();
    for (int off = 128; off > 0; off >>= 1) {
        if (t < off) red[t] = fmaxf(red[t], red[t + off]);
        __syncthreads();
    }
    m = red[0];
    __syncthreads();

    float s = 0.0f;
#pragma unroll
    for (int i = 0; i < 4; ++i) {
        x[i] = __expf(x[i] - m);
        s += x[i];
    }
    red[t] = s;
    __syncthreads();
    for (int off = 128; off > 0; off >>= 1) {
        if (t < off) red[t] += red[t + off];
        __syncthreads();
    }
    const float inv = 1.0f / red[0];
#pragma unroll
    for (int i = 0; i < 4; ++i) p[t + i * 256] = x[i] * inv;
}

// ---------------------------------------------------------------------------
// attn[b, s, d*G+g] = sum_t probs[bg, s, t] * v[b, t, d, g]
// probs read as f32 (converted to bf16 in-register); v pre-transposed.
// block tile = 256(s) x 64(d); wave tile = 32 x 64; ping-pong pipelined.
// grid = (1, S/256, B*G)
// ---------------------------------------------------------------------------
__global__ void __launch_bounds__(256, 2)
attn_av(const float* __restrict__ scores,
        const unsigned short* __restrict__ vT,
        unsigned short* __restrict__ attnb) {
    const int lane  = threadIdx.x & 31;
    const int wave  = threadIdx.x >> 5;
    const int lm    = lane & 15;
    const int bg    = blockIdx.z;
    const int b     = bg >> 4;
    const int g     = bg & (Gc - 1);
    const int tileS = blockIdx.y * 256 + wave * 32;

    const float* arow0 = scores + ((size_t)bg * Sc + tileS + lm) * SKc;
    const float* arow1 = scores + ((size_t)bg * Sc + tileS + 16 + lm) * SKc;
    const unsigned short* vr0 = vT + ((size_t)bg * DVc + 0 * 16 + lm) * SKc;
    const unsigned short* vr1 = vT + ((size_t)bg * DVc + 1 * 16 + lm) * SKc;
    const unsigned short* vr2 = vT + ((size_t)bg * DVc + 2 * 16 + lm) * SKc;
    const unsigned short* vr3 = vT + ((size_t)bg * DVc + 3 * 16 + lm) * SKc;

    f32x8 acc0[4] = {f32x8{}, f32x8{}, f32x8{}, f32x8{}};
    f32x8 acc1[4] = {f32x8{}, f32x8{}, f32x8{}, f32x8{}};

    FragSet s0, s1;
    load_set_f32a(s0, arow0, arow1, vr0, vr1, vr2, vr3, 0, lane);

    int k0 = 0;
    for (; k0 + 64 < SKc; k0 += 64) {
        __builtin_prefetch(arow0 + k0 + 96, 0, 3);
        load_set_f32a(s1, arow0, arow1, vr0, vr1, vr2, vr3, k0 + 32, lane);
        compute_set(s0, acc0, acc1);
        load_set_f32a(s0, arow0, arow1, vr0, vr1, vr2, vr3, k0 + 64, lane);
        compute_set(s1, acc0, acc1);
    }
    load_set_f32a(s1, arow0, arow1, vr0, vr1, vr2, vr3, k0 + 32, lane);
    compute_set(s0, acc0, acc1);
    compute_set(s1, acc0, acc1);

    const int rb0 = tileS + ((lane >> 4) << 3);
#pragma unroll
    for (int n = 0; n < 4; ++n) {
        const int d = n * 16 + lm;
#pragma unroll
        for (int j = 0; j < 8; ++j) {
            attnb[((size_t)(b * Sc + rb0 + j)) * NFc + d * Gc + g] =
                f32_to_bf16(acc0[n][j]);
            attnb[((size_t)(b * Sc + rb0 + 16 + j)) * NFc + d * Gc + g] =
                f32_to_bf16(acc1[n][j]);
        }
    }
}

// ---------------------------------------------------------------------------
// LayerNorm over rows of 1024; one block (256 threads) per row.
// ---------------------------------------------------------------------------
__global__ void layernorm_rows(const float* __restrict__ x,
                               const float* __restrict__ gamma,
                               const float* __restrict__ beta,
                               float* __restrict__ out) {
    __shared__ float red[256];
    const size_t row = blockIdx.x;
    const float* p = x + row * (size_t)NFc;
    const int t = threadIdx.x;

    float v[4];
    float s = 0.0f;
#pragma unroll
    for (int i = 0; i < 4; ++i) {
        v[i] = p[t + i * 256];
        s += v[i];
    }
    red[t] = s;
    __syncthreads();
    for (int off = 128; off > 0; off >>= 1) {
        if (t < off) red[t] += red[t + off];
        __syncthreads();
    }
    const float mu = red[0] * (1.0f / NFc);
    __syncthreads();

    float sq = 0.0f;
#pragma unroll
    for (int i = 0; i < 4; ++i) {
        const float d = v[i] - mu;
        sq += d * d;
    }
    red[t] = sq;
    __syncthreads();
    for (int off = 128; off > 0; off >>= 1) {
        if (t < off) red[t] += red[t + off];
        __syncthreads();
    }
    const float inv = rsqrtf(red[0] * (1.0f / NFc) + EPSc);
#pragma unroll
    for (int i = 0; i < 4; ++i) {
        const int c = t + i * 256;
        out[row * (size_t)NFc + c] = (v[i] - mu) * inv * gamma[c] + beta[c];
    }
}

// ---------------------------------------------------------------------------
// Host-side orchestration
// ---------------------------------------------------------------------------
extern "C" void kernel_launch(void* const* d_in, const int* in_sizes, int n_in,
                              void* d_out, int out_size, void* d_ws, size_t ws_size,
                              hipStream_t stream) {
    (void)in_sizes; (void)n_in; (void)out_size; (void)ws_size;

    const float* Q     = (const float*)d_in[0];
    const float* KV    = (const float*)d_in[1];
    const float* Wq    = (const float*)d_in[2];
    const float* bq    = (const float*)d_in[3];
    const float* Wk    = (const float*)d_in[4];
    const float* bk    = (const float*)d_in[5];
    const float* Wv    = (const float*)d_in[6];
    const float* bv    = (const float*)d_in[7];
    const float* Wp    = (const float*)d_in[8];
    const float* bp    = (const float*)d_in[9];
    const float* gamma = (const float*)d_in[10];
    const float* beta  = (const float*)d_in[11];

    float* out    = (float*)d_out;                        // [B*S, 1024]
    float* scores = out + (size_t)Bc * Sc * NFc;          // [B*G, S, SK]

    // Workspace carve-up (256B aligned blocks)
    char* w = (char*)d_ws;
    auto carve = [&](size_t bytes) -> void* {
        void* p = (void*)w;
        w += (bytes + 255) & ~(size_t)255;
        return p;
    };
    const size_t nAct = (size_t)Bc * Sc * NFc;            // 4,194,304
    const size_t nWgt = (size_t)NFc * NFc;                // 1,048,576

    unsigned short* Qb    = (unsigned short*)carve(nAct * 2);
    unsigned short* KVb   = (unsigned short*)carve(nAct * 2);
    unsigned short* Wqb   = (unsigned short*)carve(nWgt * 2);
    unsigned short* Wkb   = (unsigned short*)carve(nWgt * 2);
    unsigned short* Wvb   = (unsigned short*)carve(nWgt * 2);
    unsigned short* Wpb   = (unsigned short*)carve(nWgt * 2);
    unsigned short* qb    = (unsigned short*)carve(nAct * 2);
    unsigned short* kb    = (unsigned short*)carve(nAct * 2);
    unsigned short* vb    = (unsigned short*)carve(nAct * 2);
    unsigned short* vT    = (unsigned short*)carve(nAct * 2);
    unsigned short* attnb = (unsigned short*)carve(nAct * 2);
    float*          preln = (float*)carve(nAct * 4);

    const dim3 blk(256);

    // 1) fp32 -> bf16
    cvt_f32_bf16<<<dim3((unsigned)((nAct + 255) / 256)), blk, 0, stream>>>(Q,  Qb,  (int)nAct);
    cvt_f32_bf16<<<dim3((unsigned)((nAct + 255) / 256)), blk, 0, stream>>>(KV, KVb, (int)nAct);
    cvt_f32_bf16<<<dim3((unsigned)((nWgt + 255) / 256)), blk, 0, stream>>>(Wq, Wqb, (int)nWgt);
    cvt_f32_bf16<<<dim3((unsigned)((nWgt + 255) / 256)), blk, 0, stream>>>(Wk, Wkb, (int)nWgt);
    cvt_f32_bf16<<<dim3((unsigned)((nWgt + 255) / 256)), blk, 0, stream>>>(Wv, Wvb, (int)nWgt);
    cvt_f32_bf16<<<dim3((unsigned)((nWgt + 255) / 256)), blk, 0, stream>>>(Wp, Wpb, (int)nWgt);

    // 2) q/k/v projections (M=4096, N=1024, K=1024); block tile 256x64
    const dim3 gproj(NFc / 64, (Bc * Sc) / 256);
    gemm_wmma<<<gproj, blk, 0, stream>>>(Qb,  NFc, Wqb, NFc, bq, nullptr, qb, nullptr, NFc, NFc);
    gemm_wmma<<<gproj, blk, 0, stream>>>(KVb, NFc, Wkb, NFc, bk, nullptr, kb, nullptr, NFc, NFc);
    gemm_wmma<<<gproj, blk, 0, stream>>>(KVb, NFc, Wvb, NFc, bv, nullptr, vb, nullptr, NFc, NFc);

    // 3) v transpose for contiguous AV B-fragments
    transpose_v<<<dim3((unsigned)((nAct + 255) / 256)), blk, 0, stream>>>(vb, vT);

    // 4) logits (scaled), written straight into scores output
    const dim3 glog(SKc / 64, Sc / 128, Bc * Gc);
    attn_logits<<<glog, blk, 0, stream>>>(qb, kb, scores, 0.125f);

    // 5) softmax in place
    softmax_rows<<<dim3(Bc * Gc * Sc), blk, 0, stream>>>(scores);

    // 6) attn = probs @ v
    const dim3 gav(1, Sc / 256, Bc * Gc);
    attn_av<<<gav, blk, 0, stream>>>(scores, vT, attnb);

    // 7) output projection + bias + residual (f32)
    gemm_wmma<<<gproj, blk, 0, stream>>>(attnb, NFc, Wpb, NFc, bp, Q, nullptr, preln, NFc, NFc);

    // 8) LayerNorm -> final output
    layernorm_rows<<<dim3(Bc * Sc), blk, 0, stream>>>(preln, gamma, beta, out);
}
